// Net_47270410060342
// MI455X (gfx1250) — matrix operand order repaired and checked by reading, not compiled, not run on hardware
//
#include <hip/hip_runtime.h>
#include <hip/hip_bf16.h>
#include <cfloat>

typedef __attribute__((ext_vector_type(16))) _Float16 v16h;
typedef __attribute__((ext_vector_type(8)))  _Float16 v8h;
typedef __attribute__((ext_vector_type(8)))  float    v8f;

#define BN_EPS 1e-5f
#define NPTS   8192
#define M1     4096
#define M2     1024
#define QN     65536
#define KNB    128

// ---------------- WMMA helpers ----------------
__device__ __forceinline__ v8f wmma_f16(v16h a, v16h b, v8f c) {
  return __builtin_amdgcn_wmma_f32_16x16x32_f16(false, a, false, b, (short)0, c,
                                                false, false);
}

// A fragment: 16x32 tile, rows rowBase..rowBase+15 of buffer with row stride Ci halfs.
__device__ __forceinline__ v16h ld_afrag(const _Float16* buf, int rowBase, int Ci,
                                         int kt, int lane) {
  int row = rowBase + (lane & 15);
  int kb  = kt * 32 + ((lane >> 4) << 3);
  const _Float16* p = buf + (size_t)row * Ci + kb;
  v8h lo = *(const v8h*)p;
  v8h hi = *(const v8h*)(p + 16);
  v16h a;
#pragma unroll
  for (int i = 0; i < 8; ++i) { a[i] = lo[i]; a[i + 8] = hi[i]; }
  return a;
}

// B fragment from transposed weights Wt (N x Kpad, f16): two 8-half runs per lane (k = 8h+i, 16+8h+i).
__device__ __forceinline__ v16h ld_bfrag(const _Float16* Wt, int Kpad, int ncol0,
                                         int kt, int lane) {
  int n  = ncol0 + (lane & 15);
  int kb = kt * 32 + ((lane >> 4) << 3);
  const _Float16* p = Wt + (size_t)n * Kpad + kb;
  v8h lo = *(const v8h*)p;
  v8h hi = *(const v8h*)(p + 16);
  v16h b;
#pragma unroll
  for (int i = 0; i < 8; ++i) { b[i] = lo[i]; b[i + 8] = hi[i]; }
  return b;
}

// 16-byte async copy global -> LDS, tracked by ASYNCcnt (CDNA5 TDM-class path).
__device__ __forceinline__ void async_g2l_b128(const _Float16* ldsDst,
                                               const _Float16* gSrc) {
  unsigned lds_off = (unsigned)(size_t)(const void*)ldsDst;  // addr[31:0] = LDS offset
  unsigned long long ga = (unsigned long long)(size_t)gSrc;
  asm volatile("global_load_async_to_lds_b128 %0, %1, off"
               :
               : "v"(lds_off), "v"(ga)
               : "memory");
}
__device__ __forceinline__ void wait_asynccnt0() {
  asm volatile("s_wait_asynccnt 0x0" ::: "memory");
}

// ---------------- weight prep: transpose to NxKpad f16, fold BN ----------------
__global__ void prep_wt(const float* __restrict__ W, const float* __restrict__ b,
                        const float* __restrict__ g, const float* __restrict__ be,
                        const float* __restrict__ m, const float* __restrict__ v,
                        int Kdim, int N, int Kpad,
                        _Float16* __restrict__ Wt, float* __restrict__ bf) {
  int i = blockIdx.x * 256 + threadIdx.x;
  if (i >= N * Kpad) return;
  int n = i / Kpad, kp = i - n * Kpad;
  float s = 1.f, t = 0.f;
  if (g) { s = g[n] * rsqrtf(v[n] + BN_EPS); t = be[n] - m[n] * s; }
  float wv = (kp < Kdim) ? W[(size_t)kp * N + n] * s : 0.f;
  Wt[(size_t)n * Kpad + kp] = (_Float16)wv;
  if (kp == 0) bf[n] = b[n] * s + t;
}

// ---------------- FPS: serial scan, single workgroup ----------------
__global__ void fps_kernel(const float* __restrict__ pos, int n, int m,
                           int* __restrict__ idx) {
  __shared__ float sd[NPTS];
  __shared__ float rv[1024];
  __shared__ int   ri[1024];
  __shared__ float spt[3];
  int tid = threadIdx.x, nt = blockDim.x;
  float p0x = pos[0], p0y = pos[1], p0z = pos[2];
  for (int i = tid; i < n; i += nt) {
    float dx = pos[i * 3] - p0x, dy = pos[i * 3 + 1] - p0y, dz = pos[i * 3 + 2] - p0z;
    sd[i] = dx * dx + dy * dy + dz * dz;
  }
  if (tid == 0) idx[0] = 0;
  __syncthreads();
  for (int s = 1; s < m; ++s) {
    float bv = -FLT_MAX; int bi = 0x7fffffff;
    for (int i = tid; i < n; i += nt) {
      float d = sd[i];
      if (d > bv || (d == bv && i < bi)) { bv = d; bi = i; }
    }
    rv[tid] = bv; ri[tid] = bi;
    __syncthreads();
    for (int off = nt >> 1; off > 0; off >>= 1) {
      if (tid < off) {
        float ov = rv[tid + off]; int oi = ri[tid + off];
        if (ov > rv[tid] || (ov == rv[tid] && oi < ri[tid])) { rv[tid] = ov; ri[tid] = oi; }
      }
      __syncthreads();
    }
    int sel = ri[0];
    if (tid == 0) {
      idx[s] = sel;
      spt[0] = pos[sel * 3]; spt[1] = pos[sel * 3 + 1]; spt[2] = pos[sel * 3 + 2];
    }
    __syncthreads();
    float qx = spt[0], qy = spt[1], qz = spt[2];
    for (int i = tid; i < n; i += nt) {
      float dx = pos[i * 3] - qx, dy = pos[i * 3 + 1] - qy, dz = pos[i * 3 + 2] - qz;
      float nd = dx * dx + dy * dy + dz * dz;
      if (nd < sd[i]) sd[i] = nd;
    }
    __syncthreads();
  }
}

// ---------------- radius-limited top-K (K=128), d^2 kept in LDS ----------------
__global__ void radius_topk(const float* __restrict__ src, int n,
                            const float* __restrict__ dst, float r2,
                            int* __restrict__ nidx) {
  __shared__ float sd[NPTS];
  __shared__ float rv[256];
  __shared__ int   ri[256];
  int d = blockIdx.x, tid = threadIdx.x;
  float dx0 = dst[d * 3], dy0 = dst[d * 3 + 1], dz0 = dst[d * 3 + 2];
  for (int i = tid; i < n; i += 256) {
    float dx = src[i * 3] - dx0, dy = src[i * 3 + 1] - dy0, dz = src[i * 3 + 2] - dz0;
    sd[i] = dx * dx + dy * dy + dz * dz;
  }
  __syncthreads();
  for (int k = 0; k < KNB; ++k) {
    float bv = FLT_MAX; int bi = 0x7fffffff;
    for (int i = tid; i < n; i += 256) {
      float t = sd[i];
      if (t < bv || (t == bv && i < bi)) { bv = t; bi = i; }
    }
    rv[tid] = bv; ri[tid] = bi;
    __syncthreads();
    for (int off = 128; off > 0; off >>= 1) {
      if (tid < off) {
        float ov = rv[tid + off]; int oi = ri[tid + off];
        if (ov < rv[tid] || (ov == rv[tid] && oi < ri[tid])) { rv[tid] = ov; ri[tid] = oi; }
      }
      __syncthreads();
    }
    if (tid == 0) {
      nidx[(size_t)d * KNB + k] = (rv[0] <= r2) ? ri[0] : -1;
      sd[ri[0]] = FLT_MAX;
    }
    __syncthreads();
  }
}

__global__ void gather_pos(const float* __restrict__ pos, const int* __restrict__ idx,
                           float* __restrict__ out, int m) {
  int i = blockIdx.x * 256 + threadIdx.x;
  if (i < m * 3) { int r = i / 3, c = i - 3 * r; out[i] = pos[idx[r] * 3 + c]; }
}

// -------- one MLP layer inside LDS ping-pong buffers (per-wave 16 rows) --------
__device__ __forceinline__ void mlp_layer(const _Float16* cur, _Float16* nxt,
                                          const _Float16* Wt, const float* bf,
                                          int Ci, int Co, int relu, int lane, int wv) {
  int rowBase = wv * 16;
  for (int nt = 0; nt < (Co >> 4); ++nt) {
    int ncol = nt * 16 + (lane & 15);
    float bv = bf[ncol];
    v8f acc;
#pragma unroll
    for (int r = 0; r < 8; ++r) acc[r] = bv;
    for (int kt = 0; kt < (Ci >> 5); ++kt) {
      v16h a = ld_afrag(cur, rowBase, Ci, kt, lane);
      v16h b = ld_bfrag(Wt, Ci, nt * 16, kt, lane);
      acc = wmma_f16(a, b, acc);
    }
    int rb = rowBase + ((lane >> 4) << 3);
#pragma unroll
    for (int r = 0; r < 8; ++r) {
      float v = acc[r];
      if (relu) v = fmaxf(v, 0.f);
      nxt[(size_t)(rb + r) * Co + ncol] = (_Float16)v;
    }
  }
}

// ------ fused SA module: gather -> 3-layer WMMA MLP in LDS -> masked maxpool ------
__global__ void sa_fused(const float* __restrict__ srcx_f32,
                         const _Float16* __restrict__ srcx_f16, int xC,
                         const float* __restrict__ src_pos,
                         const float* __restrict__ dst_pos,
                         const int* __restrict__ nidx,
                         const _Float16* __restrict__ Wt0, const _Float16* __restrict__ Wt1,
                         const _Float16* __restrict__ Wt2,
                         const float* __restrict__ bf0, const float* __restrict__ bf1,
                         const float* __restrict__ bf2,
                         int C0, int C1, int C2, int C3, int Cmax,
                         _Float16* __restrict__ out_f16) {
  extern __shared__ _Float16 smem[];
  __shared__ int smask[KNB];
  _Float16* bufA = smem;
  _Float16* bufB = smem + (size_t)KNB * Cmax;
  int tid = threadIdx.x, lane = tid & 31, wv = tid >> 5;
  int dst = blockIdx.x;

  if (tid < KNB) smask[tid] = nidx[(size_t)dst * KNB + tid];
  __syncthreads();
  float dpx = dst_pos[dst * 3], dpy = dst_pos[dst * 3 + 1], dpz = dst_pos[dst * 3 + 2];
  for (int e = tid; e < KNB * C0; e += 256) {
    int j = e / C0, c = e - j * C0;
    int id = smask[j];
    float val = 0.f;
    if (id >= 0) {
      if (c < xC)
        val = srcx_f32 ? srcx_f32[(size_t)id * xC + c] : (float)srcx_f16[(size_t)id * xC + c];
      else if (c < xC + 3) {
        int ax = c - xC;
        val = src_pos[id * 3 + ax] - (ax == 0 ? dpx : (ax == 1 ? dpy : dpz));
      }
    }
    bufA[e] = (_Float16)val;
  }
  __syncthreads();

  mlp_layer(bufA, bufB, Wt0, bf0, C0, C1, 1, lane, wv);
  __syncthreads();
  mlp_layer(bufB, bufA, Wt1, bf1, C1, C2, 1, lane, wv);
  __syncthreads();
  mlp_layer(bufA, bufB, Wt2, bf2, C2, C3, 0, lane, wv);
  __syncthreads();

  for (int c = tid; c < C3; c += 256) {
    float best = 0.f; int any = 0;
    for (int j = 0; j < KNB; ++j) {
      if (smask[j] >= 0) {
        float v = (float)bufB[(size_t)j * C3 + c];
        best = any ? fmaxf(best, v) : v;
        any = 1;
      }
    }
    out_f16[(size_t)dst * C3 + c] = (_Float16)best;
  }
}

// ---------- generic GEMM: out = act(A[MxKpad] @ Wt^T + b), 128-row blocks ----------
// A panel (128 x Kpad) staged in LDS once via async global->LDS copies (ASYNCcnt),
// weight panel pre-warmed with global_prefetch; fragments then served from LDS.
__global__ void gemm_f16(const _Float16* __restrict__ A, const _Float16* __restrict__ Wt,
                         const float* __restrict__ bias,
                         _Float16* __restrict__ outH, float* __restrict__ outF,
                         int M, int Kpad, int N, int relu) {
  extern __shared__ _Float16 As[];  // 128 * Kpad halfs
  int tid = threadIdx.x, lane = tid & 31, wv = tid >> 5;
  int row0g = blockIdx.x * 128;

  // async-stage the A panel (16B granules)
  int chunksPerRow = Kpad >> 3;
  int totalChunks  = 128 * chunksPerRow;
  for (int c = tid; c < totalChunks; c += 256) {
    int row = c / chunksPerRow, ch = c - row * chunksPerRow;
    int grow = row0g + row;
    if (grow >= M) grow = M - 1;
    async_g2l_b128(As + (size_t)row * Kpad + ch * 8,
                   A + (size_t)grow * Kpad + ch * 8);
  }
  // pre-warm the weight panel into cache while async copies are in flight
  {
    const char* wbytes = (const char*)Wt;
    size_t wsz = (size_t)N * Kpad * sizeof(_Float16);
    for (size_t p = (size_t)tid * 128; p < wsz; p += 256 * 128)
      __builtin_prefetch(wbytes + p, 0, 1);
  }
  wait_asynccnt0();
  __syncthreads();

  int rowBase = wv * 16;
  for (int nt = 0; nt < (N >> 4); ++nt) {
    int ncol = nt * 16 + (lane & 15);
    float bv = bias[ncol];
    v8f acc;
#pragma unroll
    for (int r = 0; r < 8; ++r) acc[r] = bv;
    for (int kt = 0; kt < (Kpad >> 5); ++kt) {
      v16h a = ld_afrag(As, rowBase, Kpad, kt, lane);
      v16h b = ld_bfrag(Wt, Kpad, nt * 16, kt, lane);
      acc = wmma_f16(a, b, acc);
    }
    int rb = row0g + rowBase + ((lane >> 4) << 3);
#pragma unroll
    for (int r = 0; r < 8; ++r) {
      int rr = rb + r;
      if (rr < M) {
        float v = acc[r];
        if (relu) v = fmaxf(v, 0.f);
        if (outH) outH[(size_t)rr * N + ncol] = (_Float16)v;
        if (outF) outF[(size_t)rr * N + ncol] = v;
      }
    }
  }
}

// ---------------- small VALU kernels ----------------
__global__ void concat_a3(const _Float16* __restrict__ x2h, const float* __restrict__ pos2,
                          _Float16* __restrict__ A3) {
  int i = blockIdx.x * 256 + threadIdx.x;
  if (i >= M2 * 288) return;
  int r = i / 288, c = i - r * 288;
  float v = 0.f;
  if (c < 256) v = (float)x2h[(size_t)r * 256 + c];
  else if (c < 259) v = pos2[r * 3 + (c - 256)];
  A3[i] = (_Float16)v;
}

__global__ void colmax(const float* __restrict__ F, int M, int N, float* __restrict__ out) {
  int c = blockIdx.x * 256 + threadIdx.x;
  if (c >= N) return;
  float m = -FLT_MAX;
  for (int r = 0; r < M; ++r) m = fmaxf(m, F[(size_t)r * N + c]);
  out[c] = m;
}

__global__ void rowvec_linear(const float* __restrict__ x, const float* __restrict__ W,
                              const float* __restrict__ b, float* __restrict__ out,
                              int Kdim, int N) {
  int n = blockIdx.x * 256 + threadIdx.x;
  if (n >= N) return;
  float a = b[n];
  for (int k = 0; k < Kdim; ++k) a += x[k] * W[(size_t)k * N + n];
  out[n] = a;
}

__global__ void z_kernel(const float* __restrict__ mu, const float* __restrict__ lv,
                         const float* __restrict__ eps, float* __restrict__ z) {
  int i = blockIdx.x * 256 + threadIdx.x;
  if (i < 512) z[i] = mu[i] + eps[i] * expf(0.5f * lv[i]);
}

// decoder layer 1: relu(qp @ W1[512:515,:] + (z @ W1[:512,:] + b1))  (K=3 -> VALU)
__global__ void dec1_kernel(const float* __restrict__ qp, const float* __restrict__ W1,
                            const float* __restrict__ c1, _Float16* __restrict__ H1) {
  int i = blockIdx.x * 256 + threadIdx.x;
  if (i >= QN * 256) return;
  int q = i >> 8, n = i & 255;
  float v = c1[n] + qp[q * 3] * W1[512 * 256 + n] + qp[q * 3 + 1] * W1[513 * 256 + n] +
            qp[q * 3 + 2] * W1[514 * 256 + n];
  H1[i] = (_Float16)fmaxf(v, 0.f);
}

__global__ void concat_a2(const _Float16* __restrict__ D2, const float* __restrict__ qp,
                          _Float16* __restrict__ A2) {
  int i = blockIdx.x * 256 + threadIdx.x;
  if (i >= QN * 160) return;
  int q = i / 160, c = i - q * 160;
  float v = 0.f;
  if (c < 128) v = (float)D2[(size_t)q * 128 + c];
  else if (c < 131) v = qp[q * 3 + (c - 128)];
  A2[i] = (_Float16)v;
}

__global__ void final_kernel(const _Float16* __restrict__ D3, const float* __restrict__ Wl,
                             const float* __restrict__ bl, float* __restrict__ out) {
  int q = blockIdx.x * 256 + threadIdx.x;
  if (q >= QN) return;
  float a = bl[0];
  for (int k = 0; k < 64; ++k) a += (float)D3[(size_t)q * 64 + k] * Wl[k];
  out[q] = tanhf(a);
}

// ---------------- host driver ----------------
extern "C" void kernel_launch(void* const* d_in, const int* in_sizes, int n_in,
                              void* d_out, int out_size, void* d_ws, size_t ws_size,
                              hipStream_t stream) {
  const float* x   = (const float*)d_in[0];
  const float* pos = (const float*)d_in[1];
  const float* qp  = (const float*)d_in[3];
  const float* eps = (const float*)d_in[4];

  char* base = (char*)d_ws;
  size_t off = 0;
  auto alloc = [&](size_t bytes) -> void* {
    off = (off + 255) & ~(size_t)255;
    void* p = base + off;
    off += bytes;
    return p;
  };
  auto aH = [&](size_t n) { return (_Float16*)alloc(n * 2); };
  auto aF = [&](size_t n) { return (float*)alloc(n * 4); };
  auto aI = [&](size_t n) { return (int*)alloc(n * 4); };

  // weights (f16 transposed, BN folded) + fused biases
  _Float16* s1w0 = aH(64 * 32);    float* s1b0 = aF(64);
  _Float16* s1w1 = aH(64 * 64);    float* s1b1 = aF(64);
  _Float16* s1w2 = aH(128 * 64);   float* s1b2 = aF(128);
  _Float16* s2w0 = aH(128 * 160);  float* s2b0 = aF(128);
  _Float16* s2w1 = aH(128 * 128);  float* s2b1 = aF(128);
  _Float16* s2w2 = aH(256 * 128);  float* s2b2 = aF(256);
  _Float16* s3w0 = aH(256 * 288);  float* s3b0 = aF(256);
  _Float16* s3w1 = aH(512 * 256);  float* s3b1 = aF(512);
  _Float16* s3w2 = aH(1024 * 512); float* s3b2 = aF(1024);
  _Float16* d1w1 = aH(128 * 256);  float* d1b1 = aF(128);
  _Float16* d2w0 = aH(64 * 160);   float* d2b0 = aF(64);

  int*   idx1  = aI(M1);     float* pos1 = aF(M1 * 3);  int* nidx1 = aI((size_t)M1 * KNB);
  _Float16* x1h = aH((size_t)M1 * 128);
  int*   idx2  = aI(M2);     float* pos2 = aF(M2 * 3);  int* nidx2 = aI((size_t)M2 * KNB);
  _Float16* x2h = aH((size_t)M2 * 256);
  _Float16* A3 = aH((size_t)M2 * 288);
  _Float16* B1 = aH((size_t)M2 * 256);
  _Float16* B2 = aH((size_t)M2 * 512);
  float*    B3 = aF((size_t)M2 * 1024);
  float* g  = aF(1024); float* e = aF(512); float* z = aF(512); float* c1 = aF(256);
  _Float16* H1 = aH((size_t)QN * 256);
  _Float16* D2 = aH((size_t)QN * 128);
  _Float16* A2 = aH((size_t)QN * 160);
  _Float16* D3 = aH((size_t)QN * 64);

  float* out_sdf = (float*)d_out;
  float* out_mu  = out_sdf + QN;
  float* out_lv  = out_mu + 512;

  auto prep = [&](int wi, int bi, int gi, int bei, int mi, int vi, int Kd, int Nn,
                  int Kp, _Float16* Wt, float* bf) {
    int tot = Nn * Kp;
    prep_wt<<<dim3((tot + 255) / 256), dim3(256), 0, stream>>>(
        (const float*)d_in[wi], (const float*)d_in[bi],
        gi >= 0 ? (const float*)d_in[gi] : nullptr,
        bei >= 0 ? (const float*)d_in[bei] : nullptr,
        mi >= 0 ? (const float*)d_in[mi] : nullptr,
        vi >= 0 ? (const float*)d_in[vi] : nullptr, Kd, Nn, Kp, Wt, bf);
  };
  // params pytree (sorted keys): enc(5,6) lv(7,8) mu(9,10) sa1(11..24) sa2(25..38)
  // sa3(39..52) sdf1(53..56) sdf2(57..60);  per-bn-MLP: W*,b*,be*,g*,m*,v*
  prep(11, 14, 19, 17, 21, 23,   4,   64,  32, s1w0, s1b0);
  prep(12, 15, 20, 18, 22, 24,  64,   64,  64, s1w1, s1b1);
  prep(13, 16, -1, -1, -1, -1,  64,  128,  64, s1w2, s1b2);
  prep(25, 28, 33, 31, 35, 37, 131,  128, 160, s2w0, s2b0);
  prep(26, 29, 34, 32, 36, 38, 128,  128, 128, s2w1, s2b1);
  prep(27, 30, -1, -1, -1, -1, 128,  256, 128, s2w2, s2b2);
  prep(39, 42, 47, 45, 49, 51, 259,  256, 288, s3w0, s3b0);
  prep(40, 43, 48, 46, 50, 52, 256,  512, 256, s3w1, s3b1);
  prep(41, 44, -1, -1, -1, -1, 512, 1024, 512, s3w2, s3b2);
  prep(54, 56, -1, -1, -1, -1, 256,  128, 256, d1w1, d1b1);
  prep(57, 59, -1, -1, -1, -1, 131,   64, 160, d2w0, d2b0);

  // ---- SA stage 1 ----
  fps_kernel<<<1, 1024, 0, stream>>>(pos, NPTS, M1, idx1);
  gather_pos<<<(M1 * 3 + 255) / 256, 256, 0, stream>>>(pos, idx1, pos1, M1);
  radius_topk<<<M1, 256, 0, stream>>>(pos, NPTS, pos1, 0.2f * 0.2f, nidx1);
  sa_fused<<<M1, 256, 2 * KNB * 128 * sizeof(_Float16), stream>>>(
      x, nullptr, 1, pos, pos1, nidx1, s1w0, s1w1, s1w2, s1b0, s1b1, s1b2,
      32, 64, 64, 128, 128, x1h);

  // ---- SA stage 2 ----
  fps_kernel<<<1, 1024, 0, stream>>>(pos1, M1, M2, idx2);
  gather_pos<<<(M2 * 3 + 255) / 256, 256, 0, stream>>>(pos1, idx2, pos2, M2);
  radius_topk<<<M2, 256, 0, stream>>>(pos1, M1, pos2, 0.5f * 0.5f, nidx2);
  sa_fused<<<M2, 256, 2 * KNB * 256 * sizeof(_Float16), stream>>>(
      nullptr, x1h, 128, pos1, pos2, nidx2, s2w0, s2w1, s2w2, s2b0, s2b1, s2b2,
      160, 128, 128, 256, 256, x2h);

  // ---- SA3 MLP + global max ----
  concat_a3<<<(M2 * 288 + 255) / 256, 256, 0, stream>>>(x2h, pos2, A3);
  gemm_f16<<<M2 / 128, 256, 128 * 288 * sizeof(_Float16), stream>>>(
      A3, s3w0, s3b0, B1, nullptr, M2, 288, 256, 1);
  gemm_f16<<<M2 / 128, 256, 128 * 256 * sizeof(_Float16), stream>>>(
      B1, s3w1, s3b1, B2, nullptr, M2, 256, 512, 1);
  gemm_f16<<<M2 / 128, 256, 128 * 512 * sizeof(_Float16), stream>>>(
      B2, s3w2, s3b2, nullptr, B3, M2, 512, 1024, 0);
  colmax<<<(1024 + 255) / 256, 256, 0, stream>>>(B3, M2, 1024, g);

  // ---- VAE head (M=1 -> VALU) ----
  rowvec_linear<<<2, 256, 0, stream>>>(g, (const float*)d_in[5], (const float*)d_in[6],
                                       e, 1024, 512);
  rowvec_linear<<<2, 256, 0, stream>>>(e, (const float*)d_in[9], (const float*)d_in[10],
                                       out_mu, 512, 512);
  rowvec_linear<<<2, 256, 0, stream>>>(e, (const float*)d_in[7], (const float*)d_in[8],
                                       out_lv, 512, 512);
  z_kernel<<<2, 256, 0, stream>>>(out_mu, out_lv, eps, z);

  // ---- SDF decoder ----
  rowvec_linear<<<1, 256, 0, stream>>>(z, (const float*)d_in[53], (const float*)d_in[55],
                                       c1, 512, 256);
  dec1_kernel<<<(QN * 256) / 256, 256, 0, stream>>>(qp, (const float*)d_in[53], c1, H1);
  gemm_f16<<<QN / 128, 256, 128 * 256 * sizeof(_Float16), stream>>>(
      H1, d1w1, d1b1, D2, nullptr, QN, 256, 128, 1);
  concat_a2<<<(QN * 160 + 255) / 256, 256, 0, stream>>>(D2, qp, A2);
  gemm_f16<<<QN / 128, 256, 128 * 160 * sizeof(_Float16), stream>>>(
      A2, d2w0, d2b0, D3, nullptr, QN, 160, 64, 1);
  final_kernel<<<QN / 256, 256, 0, stream>>>(D3, (const float*)d_in[58],
                                             (const float*)d_in[60], out_sdf);
}
